// SelfAttention_5214090297832
// MI455X (gfx1250) — matrix operand order repaired and checked by reading, not compiled, not run on hardware
//
#include <hip/hip_runtime.h>

#define B_ 8
#define C_ 64
#define N_ 4096
#define DH 32

typedef __attribute__((ext_vector_type(16))) _Float16 v16h;
typedef __attribute__((ext_vector_type(8)))  _Float16 v8h;
typedef __attribute__((ext_vector_type(8)))  float    v8f;

__device__ __forceinline__ v8f wmma_f16(v16h a, v16h b, v8f c) {
  // D = A(16x32 f16) x B(32x16 f16) + C(16x16 f32)
  return __builtin_amdgcn_wmma_f32_16x16x32_f16(false, a, false, b, (short)0, c,
                                                false, false);
}

// base-2 exponential -> single v_exp_f32
__device__ __forceinline__ float fast_exp2(float x) {
  return __builtin_amdgcn_exp2f(x);
}

// assemble a v16h operand from two contiguous 16-byte chunks
__device__ __forceinline__ v16h ld_op(const _Float16* p0, const _Float16* p1) {
  union { v16h v; v8h p[2]; } u;
  u.p[0] = *(const v8h*)p0;
  u.p[1] = *(const v8h*)p1;
  return u.v;
}

__device__ __forceinline__ void lds_fence() {
  asm volatile("s_wait_dscnt 0" ::: "memory");
}
__device__ __forceinline__ void sched_fence() {
  asm volatile("" ::: "memory");
}

// raw v_permlane16_b32 on a float (no tied-mov, no canonicalize)
__device__ __forceinline__ float permlane16_f(float v, int sel0, int sel1) {
  float d;
  asm("v_permlane16_b32 %0, %1, %2, %3"
      : "=v"(d)
      : "v"(v), "s"(sel0), "s"(sel1));
  return d;
}
// xor-butterfly max over each 16-lane row (columns of one half-wave)
__device__ __forceinline__ float rowmax16(float v) {
  v = fmaxf(v, permlane16_f(v, 0x67452301, (int)0xEFCDAB89));  // xor 1
  v = fmaxf(v, permlane16_f(v, 0x54761032, (int)0xDCFE98BA));  // xor 2
  v = fmaxf(v, permlane16_f(v, 0x32107654, (int)0xBA98FEDC));  // xor 4
  v = fmaxf(v, permlane16_f(v, (int)0xFEDCBA98, 0x76543210));  // xor 8
  return v;
}

// online softmax update for one 16-row half: returns "max moved" flag,
// writes P = 2^(s - mnew) into smrow ( [16][64] f16, row m = r + 8*hlf )
__device__ __forceinline__ int softmax_half(const v8f* s, float* mrow,
                                            float* sc, _Float16* smrow,
                                            int col, int hlf) {
  int anyup = 0;
#pragma unroll
  for (int r = 0; r < 8; ++r) {
    float mx = fmaxf(fmaxf(s[0][r], s[1][r]), fmaxf(s[2][r], s[3][r]));
    mx = rowmax16(mx);
    const float mo = mrow[r];
    const float mnew = fmaxf(mo, mx);
    mrow[r] = mnew;
    sc[r] = fast_exp2(mo - mnew);
    anyup |= (mnew > mo);
    _Float16* smr = smrow + (r + 8 * hlf) * 64 + col;
    smr[0]  = (_Float16)fast_exp2(s[0][r] - mnew);
    smr[16] = (_Float16)fast_exp2(s[1][r] - mnew);
    smr[32] = (_Float16)fast_exp2(s[2][r] - mnew);
    smr[48] = (_Float16)fast_exp2(s[3][r] - mnew);
  }
  return anyup;
}

// ---------------------------------------------------------------------------
// Projection: x[B,C,N] f32 -> Q[B,N,32] f16 (pre-scaled by log2(e)),
//             K[B,N,32] f16, V[B,C,N] f16
// ---------------------------------------------------------------------------
__global__ __launch_bounds__(256) void proj_kernel(
    const float* __restrict__ x,
    const float* __restrict__ wq, const float* __restrict__ bq,
    const float* __restrict__ wk, const float* __restrict__ bk,
    const float* __restrict__ wv, const float* __restrict__ bv,
    _Float16* __restrict__ Qh, _Float16* __restrict__ Kh,
    _Float16* __restrict__ Vh) {
  const int idx = blockIdx.x * 256 + threadIdx.x;   // b*N + n
  const int b = idx >> 12;
  const int n = idx & (N_ - 1);
  const float* xcol = x + (b * C_) * N_ + n;
  float xc[C_];
#pragma unroll
  for (int c = 0; c < C_; ++c) xc[c] = xcol[c * N_];

  _Float16* qo = Qh + idx * DH;
  _Float16* ko = Kh + idx * DH;
#pragma unroll 4
  for (int o = 0; o < DH; ++o) {
    float qa = bq[o], ka = bk[o];
#pragma unroll
    for (int c = 0; c < C_; ++c) {
      qa = fmaf(wq[o * C_ + c], xc[c], qa);
      ka = fmaf(wk[o * C_ + c], xc[c], ka);
    }
    qo[o] = (_Float16)(qa * 1.44269504f);  // logits become base-2
    ko[o] = (_Float16)ka;
  }
#pragma unroll 4
  for (int o = 0; o < C_; ++o) {
    float va = bv[o];
#pragma unroll
    for (int c = 0; c < C_; ++c) va = fmaf(wv[o * C_ + c], xc[c], va);
    Vh[(b * C_ + o) * N_ + n] = (_Float16)va;
  }
}

__global__ __launch_bounds__(256) void cvt_wz_kernel(const float* __restrict__ wz,
                                                     _Float16* __restrict__ wzh) {
  int i = blockIdx.x * 256 + threadIdx.x;
  if (i < C_ * C_) wzh[i] = (_Float16)wz[i];
}

// ---------------------------------------------------------------------------
// Fused flash attention + output projection + residual.
// 1 wave = 32 query rows x 64 keys per iteration; 4 waves / block.
// K/V operands are reused by both 16-row halves -> 2x arithmetic intensity.
// ---------------------------------------------------------------------------
__global__ __launch_bounds__(128) void attn_kernel(
    const _Float16* __restrict__ Q, const _Float16* __restrict__ K,
    const _Float16* __restrict__ V, const _Float16* __restrict__ WZ,
    const float* __restrict__ bz, const float* __restrict__ hres,
    float* __restrict__ out) {
  __shared__ __align__(16) _Float16 sbuf[4][32 * 64];  // per-wave slice

  const int lane = threadIdx.x & 31;
  const int wid = threadIdx.x >> 5;
  const int col = lane & 15;  // n / m index within a 16-wide tile
  const int hlf = lane >> 4;  // which half of the wave
  const int b = blockIdx.x >> 5;                       // 32 blocks per batch
  const int rowBase = ((blockIdx.x & 31) << 7) + (wid << 5);
  _Float16* sm = sbuf[wid];

  const _Float16* Qb = Q + (b * N_) * DH;
  const _Float16* Kb = K + (b * N_) * DH;
  const _Float16* Vb = V + (b * C_) * N_;

  // two A operands: Q rows rowBase..+15 (half A) and rowBase+16..+31 (half B)
  const _Float16* qr0 = Qb + (rowBase + col) * DH + 8 * hlf;
  const _Float16* qr1 = qr0 + 16 * DH;
  const v16h qa0 = ld_op(qr0, qr0 + 16);
  const v16h qa1 = ld_op(qr1, qr1 + 16);

  // all-ones B operand: row-sum of P via WMMA (lands in C-layout like acc)
  v16h ones;
#pragma unroll
  for (int i = 0; i < 16; ++i) ones[i] = (_Float16)1.0f;

  v8f accA[4] = {v8f{}, v8f{}, v8f{}, v8f{}};
  v8f accB[4] = {v8f{}, v8f{}, v8f{}, v8f{}};
  v8f laccA = {}, laccB = {};
  float mrowA[8], mrowB[8];
#pragma unroll
  for (int r = 0; r < 8; ++r) { mrowA[r] = -3.0e38f; mrowB[r] = -3.0e38f; }

  for (int j = 0; j < N_; j += 64) {
    // S = Q x K^T : each K tile feeds both row halves
    v8f sA[4], sB[4];
#pragma unroll
    for (int q = 0; q < 4; ++q) {
      const _Float16* kr = Kb + (j + 16 * q + col) * DH + 8 * hlf;
      v16h kb = ld_op(kr, kr + 16);
      sA[q] = wmma_f16(qa0, kb, v8f{});
      sB[q] = wmma_f16(qa1, kb, v8f{});
    }
    // prefetch V^T B operands (4 channel tiles x 2 key chunks)
    v16h vb[8];
#pragma unroll
    for (int t = 0; t < 4; ++t) {
      const _Float16* vr = Vb + (t * 16 + col) * N_ + j + 8 * hlf;
      vb[2 * t] = ld_op(vr, vr + 16);
      vb[2 * t + 1] = ld_op(vr + 32, vr + 48);
    }

    float scA[8], scB[8];
    int anyup = softmax_half(sA, mrowA, scA, sm, col, hlf) |
                softmax_half(sB, mrowB, scB, sm + 16 * 64, col, hlf);
    lds_fence();

    // P as A operands (16x32 each), read back transposed across lanes
    const _Float16* prA = sm + col * 64 + 8 * hlf;
    const _Float16* prB = prA + 16 * 64;
    v16h paA0 = ld_op(prA, prA + 16), paA1 = ld_op(prA + 32, prA + 48);
    v16h paB0 = ld_op(prB, prB + 16), paB1 = ld_op(prB + 32, prB + 48);
    sched_fence();

    // rescale running sums/outputs only when a max actually moved
    if (__any(anyup)) {
#pragma unroll
      for (int r = 0; r < 8; ++r) {
        laccA[r] *= scA[r];
        laccB[r] *= scB[r];
#pragma unroll
        for (int t = 0; t < 4; ++t) {
          accA[t][r] *= scA[r];
          accB[t][r] *= scB[r];
        }
      }
    }
    laccA = wmma_f16(paA0, ones, laccA);   // l += rowsum(P)
    laccA = wmma_f16(paA1, ones, laccA);
    laccB = wmma_f16(paB0, ones, laccB);
    laccB = wmma_f16(paB1, ones, laccB);
#pragma unroll
    for (int t = 0; t < 4; ++t) {          // O[m,c] += P x V^T
      accA[t] = wmma_f16(paA0, vb[2 * t], accA[t]);
      accA[t] = wmma_f16(paA1, vb[2 * t + 1], accA[t]);
      accB[t] = wmma_f16(paB0, vb[2 * t], accB[t]);
      accB[t] = wmma_f16(paB1, vb[2 * t + 1], accB[t]);
    }
  }

  // normalize and stage z[m][c] (32x64 f16) in LDS for the output projection
  float rinvA[8], rinvB[8];
#pragma unroll
  for (int r = 0; r < 8; ++r) {
    rinvA[r] = 1.0f / laccA[r];
    rinvB[r] = 1.0f / laccB[r];
  }
#pragma unroll
  for (int t = 0; t < 4; ++t)
#pragma unroll
    for (int r = 0; r < 8; ++r) {
      sm[(r + 8 * hlf) * 64 + t * 16 + col] =
          (_Float16)(accA[t][r] * rinvA[r]);
      sm[(16 + r + 8 * hlf) * 64 + t * 16 + col] =
          (_Float16)(accB[t][r] * rinvB[r]);
    }
  lds_fence();

  // out[o, m] = wz[o,:] . z[:, m] + bz[o] + h[o, m]  (both row halves)
#pragma unroll
  for (int t = 0; t < 4; ++t) {
    v8f dA = {}, dB = {};
#pragma unroll
    for (int kc = 0; kc < C_; kc += 32) {
      const _Float16* ar = WZ + (t * 16 + col) * C_ + kc + 8 * hlf;
      v16h a = ld_op(ar, ar + 16);
      const _Float16* brA = sm + col * 64 + kc + 8 * hlf;
      const _Float16* brB = brA + 16 * 64;
      dA = wmma_f16(a, ld_op(brA, brA + 16), dA);
      dB = wmma_f16(a, ld_op(brB, brB + 16), dB);
    }
#pragma unroll
    for (int r = 0; r < 8; ++r) {
      const int o = t * 16 + r + 8 * hlf;
      const int gA = (b * C_ + o) * N_ + rowBase + col;
      const int gB = gA + 16;
      out[gA] = dA[r] + bz[o] + hres[gA];
      out[gB] = dB[r] + bz[o] + hres[gB];
    }
  }
}

// ---------------------------------------------------------------------------
extern "C" void kernel_launch(void* const* d_in, const int* in_sizes, int n_in,
                              void* d_out, int out_size, void* d_ws,
                              size_t ws_size, hipStream_t stream) {
  const float* h = (const float*)d_in[0];
  const float* wq = (const float*)d_in[1];
  const float* bq = (const float*)d_in[2];
  const float* wk = (const float*)d_in[3];
  const float* bk = (const float*)d_in[4];
  const float* wv = (const float*)d_in[5];
  const float* bv = (const float*)d_in[6];
  const float* wz = (const float*)d_in[7];
  const float* bz = (const float*)d_in[8];
  float* out = (float*)d_out;

  _Float16* Qh = (_Float16*)d_ws;            // B*N*32 halfs
  _Float16* Kh = Qh + (size_t)B_ * N_ * DH;  // B*N*32 halfs
  _Float16* Vh = Kh + (size_t)B_ * N_ * DH;  // B*C*N halfs
  _Float16* WZh = Vh + (size_t)B_ * C_ * N_; // 4096 halfs

  proj_kernel<<<(B_ * N_) / 256, 256, 0, stream>>>(h, wq, bq, wk, bk, wv, bv,
                                                   Qh, Kh, Vh);
  cvt_wz_kernel<<<(C_ * C_ + 255) / 256, 256, 0, stream>>>(wz, WZh);
  attn_kernel<<<B_ * (N_ / 128), 128, 0, stream>>>(Qh, Kh, Vh, WZh, bz, h, out);
}
